// TabNetEncoder_1047972020792
// MI455X (gfx1250) — matrix operand order repaired and checked by reading, not compiled, not run on hardware
//
#include <hip/hip_runtime.h>
#include <hip/hip_bf16.h>

typedef __attribute__((ext_vector_type(16))) __bf16 v16bf;
typedef __attribute__((ext_vector_type(8)))  float  v8f;

struct __align__(16) U128 { unsigned int a, b, c, d; };
union AFrag { v16bf v; U128 q[2]; __device__ AFrag() {} };

static constexpr int   kB     = 65536;
static constexpr int   kD     = 512;
static constexpr int   kH     = 128;
static constexpr int   k2H    = 256;
static constexpr float kScale = 0.70710678118654752440f; // sqrt(0.5)
static constexpr float kGamma = 1.3f;

__device__ __forceinline__ unsigned short f2bf(float f) {
  unsigned int u = __float_as_uint(f);
  u += 0x7FFFu + ((u >> 16) & 1u);           // round-to-nearest-even
  return (unsigned short)(u >> 16);
}

// ---------------- full-batch BN (bn0): one block per feature ----------------
__global__ __launch_bounds__(256) void bn0_stats_k(const float* __restrict__ x,
                                                   float* __restrict__ mean,
                                                   float* __restrict__ rstd) {
  const int f = blockIdx.x, tid = threadIdx.x;
  float s = 0.f, s2 = 0.f;
  for (int r = tid; r < kB; r += 256) {
    float v = x[(size_t)r * kD + f];
    s += v; s2 += v * v;
  }
  __shared__ float sh[256], sh2[256];
  sh[tid] = s; sh2[tid] = s2; __syncthreads();
  for (int off = 128; off > 0; off >>= 1) {
    if (tid < off) { sh[tid] += sh[tid + off]; sh2[tid] += sh2[tid + off]; }
    __syncthreads();
  }
  if (tid == 0) {
    float m = sh[0] * (1.f / (float)kB);
    float v = sh2[0] * (1.f / (float)kB) - m * m;
    mean[f] = m; rstd[f] = rsqrtf(v + 1e-5f);
  }
}

__global__ __launch_bounds__(256) void bn0_apply_k(const float* __restrict__ x,
    const float* __restrict__ mean, const float* __restrict__ rstd,
    const float* __restrict__ g, const float* __restrict__ b,
    float* __restrict__ xn, unsigned short* __restrict__ xnb) {
  size_t i = (size_t)blockIdx.x * 256 + threadIdx.x;
  int f = (int)(i & (kD - 1));
  float v = (x[i] - mean[f]) * rstd[f] * g[f] + b[f];
  xn[i] = v; xnb[i] = f2bf(v);
}

__global__ __launch_bounds__(256) void cvt_bf16_k(const float* __restrict__ s,
                                                  unsigned short* __restrict__ d, int n) {
  int i = blockIdx.x * 256 + threadIdx.x;
  if (i < n) d[i] = f2bf(s[i]);
}

__global__ void zero_k(float* p) { p[0] = 0.f; }

// ---------------- bf16 WMMA GEMM: C[M,N] = A[M,K] * W[N,K]^T ----------------
// block = 8 waves, block tile 128x64; each wave: 16 rows x 64 cols (4 WMMAs/Kstep)
__global__ __launch_bounds__(256) void gemm_bf16_k(const unsigned short* __restrict__ A,
    const unsigned short* __restrict__ W, float* __restrict__ C, int N, int K) {
  const int lane = threadIdx.x & 31;
  const int wave = threadIdx.x >> 5;
  const int lmod = lane & 15;
  const int lgrp = lane >> 4;
  const int m0 = blockIdx.x * 128 + wave * 16;
  const int n0 = blockIdx.y * 64;
  const unsigned short* aRow = A + (size_t)(m0 + lmod) * K;

  v8f acc[4] = {};
  for (int k0 = 0; k0 < K; k0 += 32) {
    AFrag a;
    const unsigned short* ap = aRow + k0 + lgrp * 16;
    a.q[0] = *reinterpret_cast<const U128*>(ap);        // K = k0+g*16 .. +7
    a.q[1] = *reinterpret_cast<const U128*>(ap + 8);    // K = k0+g*16+8 .. +7
    __builtin_prefetch(ap + 32, 0, 1);
#pragma unroll
    for (int t = 0; t < 4; ++t) {
      const unsigned short* wp =
          W + (size_t)(n0 + t * 16 + lmod) * K + k0 + lgrp * 16;
      v16bf bv = *reinterpret_cast<const v16bf*>(wp);   // 16 contiguous K
      acc[t] = __builtin_amdgcn_wmma_f32_16x16x32_bf16(
          false, a.v, false, bv, (short)0, acc[t], false, false);
    }
  }
#pragma unroll
  for (int t = 0; t < 4; ++t) {
    float* cp = C + (size_t)(m0 + 8 * lgrp) * N + n0 + t * 16 + lmod;
#pragma unroll
    for (int r = 0; r < 8; ++r) cp[(size_t)r * N] = acc[t][r];
  }
}

// ---------------- Ghost-BN stats: per 512-row chunk, per feature ----------------
__global__ __launch_bounds__(256) void gbn_stats_k(const float* __restrict__ y,
    float* __restrict__ mean, float* __restrict__ rstd, int F) {
  const int chunk = blockIdx.x;
  const int c0 = blockIdx.y * 64;
  const int cx = threadIdx.x & 63;
  const int ry = threadIdx.x >> 6;            // 0..3
  const float* base = y + (size_t)chunk * 512 * F + c0 + cx;
  float s = 0.f, s2 = 0.f;
  for (int r = ry; r < 512; r += 4) {
    float v = base[(size_t)r * F];
    s += v; s2 += v * v;
  }
  __shared__ float sh[256], sh2[256];
  sh[threadIdx.x] = s; sh2[threadIdx.x] = s2; __syncthreads();
  if (ry == 0) {
    s  = sh[cx]  + sh[cx + 64]  + sh[cx + 128]  + sh[cx + 192];
    s2 = sh2[cx] + sh2[cx + 64] + sh2[cx + 128] + sh2[cx + 192];
    float m = s * (1.f / 512.f);
    float v = s2 * (1.f / 512.f) - m * m;
    mean[chunk * F + c0 + cx] = m;
    rstd[chunk * F + c0 + cx] = rsqrtf(v + 1e-5f);
  }
}

// ---------------- GBN-apply + GLU (+residual*scale), optional split output ----------------
// mode: hOut != null  -> write h (f32 + bf16), full 128 cols
//       hOut == null  -> split: col<64 -> relu to outRelu (if non-null); col>=64 -> att (+bf16)
__global__ __launch_bounds__(256) void glu_apply_k(const float* __restrict__ y,
    const float* __restrict__ mean, const float* __restrict__ rstd,
    const float* __restrict__ g, const float* __restrict__ b,
    const float* res, float* hOut, unsigned short* hbfOut,
    float* outRelu, float* attOut, unsigned short* attbfOut) {
  const size_t i = (size_t)blockIdx.x * 256 + threadIdx.x;   // over B*128
  const size_t row = i >> 7;
  const int col = (int)(i & 127);
  const int chunk = (int)(row >> 9);
  const int F = k2H;
  const int fu = col, fv = col + kH;
  float u = y[row * F + fu];
  float v = y[row * F + fv];
  u = (u - mean[chunk * F + fu]) * rstd[chunk * F + fu] * g[fu] + b[fu];
  v = (v - mean[chunk * F + fv]) * rstd[chunk * F + fv] * g[fv] + b[fv];
  float glu = u * (1.f / (1.f + __expf(-v)));
  float o = res ? (res[i] + glu) * kScale : glu;
  if (hOut) {
    hOut[i] = o; hbfOut[i] = f2bf(o);
  } else {
    if (col < 64) {
      if (outRelu) outRelu[row * 64 + col] = fmaxf(o, 0.f);
    } else {
      attOut[row * 64 + col - 64] = o;
      attbfOut[row * 64 + col - 64] = f2bf(o);
    }
  }
}

// ---------------- fused GBN + sparsemax + loss + prior + M*x (one wave per row) ----------------
__global__ __launch_bounds__(256) void sparsemax_k(const float* __restrict__ y,
    const float* __restrict__ mean, const float* __restrict__ rstd,
    const float* __restrict__ g, const float* __restrict__ b,
    const float* __restrict__ xn, float* prior,
    unsigned short* __restrict__ mxbf, float* mloss, int firstStep) {
  const int lane = threadIdx.x & 31;
  const int wv = threadIdx.x >> 5;
  const size_t row = (size_t)blockIdx.x * 8 + wv;
  const int F = 512;
  const int chunk = (int)(row >> 9);
  float z[16], p[16];
  float zmax = -3.0e38f;
#pragma unroll
  for (int j = 0; j < 16; ++j) {
    int c = lane + 32 * j;
    float a = (y[row * F + c] - mean[chunk * F + c]) * rstd[chunk * F + c] * g[c] + b[c];
    float pr = firstStep ? 1.f : prior[row * F + c];
    p[j] = pr; z[j] = a * pr;
    zmax = fmaxf(zmax, z[j]);
  }
#pragma unroll
  for (int off = 16; off > 0; off >>= 1) zmax = fmaxf(zmax, __shfl_xor(zmax, off, 32));
  // tau* in [zmax-1, zmax): bisection on monotone S(tau)=sum relu(z-tau)
  float lo = zmax - 1.f, hi = zmax;
  for (int it = 0; it < 26; ++it) {
    float mid = 0.5f * (lo + hi);
    float s = 0.f;
#pragma unroll
    for (int j = 0; j < 16; ++j) s += fmaxf(z[j] - mid, 0.f);
#pragma unroll
    for (int off = 16; off > 0; off >>= 1) s += __shfl_xor(s, off, 32);
    if (s > 1.f) lo = mid; else hi = mid;
  }
  // exact projection with the localized support set
  float tau0 = 0.5f * (lo + hi);
  float k = 0.f, ss = 0.f;
#pragma unroll
  for (int j = 0; j < 16; ++j)
    if (z[j] > tau0) { k += 1.f; ss += z[j]; }
#pragma unroll
  for (int off = 16; off > 0; off >>= 1) {
    k += __shfl_xor(k, off, 32); ss += __shfl_xor(ss, off, 32);
  }
  const float tau = (ss - 1.f) / fmaxf(k, 1.f);
  float lsum = 0.f;
#pragma unroll
  for (int j = 0; j < 16; ++j) {
    int c = lane + 32 * j;
    float M = fmaxf(z[j] - tau, 0.f);
    lsum += M * __logf(M + 1e-15f);
    prior[row * F + c] = (kGamma - M) * p[j];
    mxbf[row * F + c] = f2bf(M * xn[row * F + c]);
  }
#pragma unroll
  for (int off = 16; off > 0; off >>= 1) lsum += __shfl_xor(lsum, off, 32);
  if (lane == 0) atomicAdd(mloss, lsum);
}

__global__ void finalize_k(const float* mloss, float* out) {
  out[0] = mloss[0] * (1.f / ((float)kB * 3.f));
}

// ================================================================================
extern "C" void kernel_launch(void* const* d_in, const int* in_sizes, int n_in,
                              void* d_out, int out_size, void* d_ws, size_t ws_size,
                              hipStream_t stream) {
  const float* x      = (const float*)d_in[0];
  const float* bn0_g  = (const float*)d_in[1];
  const float* bn0_b  = (const float*)d_in[2];
  const float* Wsh0   = (const float*)d_in[3];   // [256][512]
  const float* Wsh1   = (const float*)d_in[4];   // [256][128]
  const float* sh_g   = (const float*)d_in[5];   // [4][2][256]
  const float* sh_b   = (const float*)d_in[6];
  const float* spec_W = (const float*)d_in[7];   // [4][2][256][128]
  const float* spec_g = (const float*)d_in[8];   // [4][2][256]
  const float* spec_b = (const float*)d_in[9];
  const float* att_W  = (const float*)d_in[10];  // [3][512][64]
  const float* att_g  = (const float*)d_in[11];  // [3][512]
  const float* att_b  = (const float*)d_in[12];
  float* out = (float*)d_out;
  (void)in_sizes; (void)n_in; (void)out_size; (void)ws_size;

  char* wp = (char*)d_ws;
  auto carve = [&](size_t bytes) -> void* {
    void* r = (void*)wp; wp += (bytes + 255) & ~(size_t)255; return r;
  };
  float*          xn    = (float*)carve((size_t)kB * kD * 4);
  unsigned short* xnb   = (unsigned short*)carve((size_t)kB * kD * 2);
  float*          y     = (float*)carve((size_t)kB * 512 * 4);
  float*          h     = (float*)carve((size_t)kB * kH * 4);
  unsigned short* hbf   = (unsigned short*)carve((size_t)kB * kH * 2);
  float*          attf  = (float*)carve((size_t)kB * 64 * 4);
  unsigned short* attbf = (unsigned short*)carve((size_t)kB * 64 * 2);
  float*          prior = (float*)carve((size_t)kB * 512 * 4);
  unsigned short* mxbf  = (unsigned short*)carve((size_t)kB * 512 * 2);
  float*          bn0m  = (float*)carve(kD * 4);
  float*          bn0r  = (float*)carve(kD * 4);
  float*          gmean = (float*)carve((size_t)128 * 512 * 4);
  float*          grstd = (float*)carve((size_t)128 * 512 * 4);
  unsigned short* wsh0b = (unsigned short*)carve((size_t)256 * 512 * 2);
  unsigned short* wsh1b = (unsigned short*)carve((size_t)256 * 128 * 2);
  unsigned short* specb = (unsigned short*)carve((size_t)4 * 2 * 256 * 128 * 2);
  unsigned short* attwb = (unsigned short*)carve((size_t)3 * 512 * 64 * 2);
  float*          mloss = (float*)carve(256);

  // ---- bn0 + weight conversion ----
  bn0_stats_k<<<kD, 256, 0, stream>>>(x, bn0m, bn0r);
  bn0_apply_k<<<(unsigned)((size_t)kB * kD / 256), 256, 0, stream>>>(
      x, bn0m, bn0r, bn0_g, bn0_b, xn, xnb);
  auto cvt = [&](const float* s, unsigned short* d, int n) {
    cvt_bf16_k<<<(n + 255) / 256, 256, 0, stream>>>(s, d, n);
  };
  cvt(Wsh0, wsh0b, 256 * 512);
  cvt(Wsh1, wsh1b, 256 * 128);
  cvt(spec_W, specb, 4 * 2 * 256 * 128);
  cvt(att_W, attwb, 3 * 512 * 64);
  zero_k<<<1, 1, 0, stream>>>(mloss);

  const unsigned gluGrid = (unsigned)((size_t)kB * kH / 256);
  // ---- feature transformer: 4 GEMM+GBN+GLU layers; last layer splits out/att ----
  auto run_ft = [&](const unsigned short* inBf, int Kin, int t, float* outRelu) {
    dim3 gg(kB / 128, k2H / 64);
    dim3 sg(kB / 512, k2H / 64);
    // layer 0: shared W0, no residual
    gemm_bf16_k<<<gg, 256, 0, stream>>>(inBf, wsh0b, y, k2H, Kin);
    gbn_stats_k<<<sg, 256, 0, stream>>>(y, gmean, grstd, k2H);
    glu_apply_k<<<gluGrid, 256, 0, stream>>>(y, gmean, grstd,
        sh_g + ((size_t)t * 2 + 0) * k2H, sh_b + ((size_t)t * 2 + 0) * k2H,
        nullptr, h, hbf, nullptr, nullptr, nullptr);
    // layer 1: shared W1, residual
    gemm_bf16_k<<<gg, 256, 0, stream>>>(hbf, wsh1b, y, k2H, kH);
    gbn_stats_k<<<sg, 256, 0, stream>>>(y, gmean, grstd, k2H);
    glu_apply_k<<<gluGrid, 256, 0, stream>>>(y, gmean, grstd,
        sh_g + ((size_t)t * 2 + 1) * k2H, sh_b + ((size_t)t * 2 + 1) * k2H,
        h, h, hbf, nullptr, nullptr, nullptr);
    // layers 2,3: step-specific, residual; layer 3 splits (relu|att)
    for (int i = 0; i < 2; ++i) {
      gemm_bf16_k<<<gg, 256, 0, stream>>>(
          hbf, specb + ((size_t)t * 2 + i) * 256 * 128, y, k2H, kH);
      gbn_stats_k<<<sg, 256, 0, stream>>>(y, gmean, grstd, k2H);
      const float* gg2 = spec_g + ((size_t)t * 2 + i) * k2H;
      const float* bb2 = spec_b + ((size_t)t * 2 + i) * k2H;
      if (i == 0)
        glu_apply_k<<<gluGrid, 256, 0, stream>>>(y, gmean, grstd, gg2, bb2,
            h, h, hbf, nullptr, nullptr, nullptr);
      else
        glu_apply_k<<<gluGrid, 256, 0, stream>>>(y, gmean, grstd, gg2, bb2,
            h, nullptr, nullptr, outRelu, attf, attbf);
    }
  };

  run_ft(xnb, kD, 0, nullptr);   // initial att = ft(x,0)[:, N_D:]

  for (int s = 0; s < 3; ++s) {
    // a = GBN(att @ att_W[s]^T): [B,64] x [512,64]^T -> [B,512]
    gemm_bf16_k<<<dim3(kB / 128, 512 / 64), 256, 0, stream>>>(
        attbf, attwb + (size_t)s * 512 * 64, y, 512, 64);
    gbn_stats_k<<<dim3(kB / 512, 512 / 64), 256, 0, stream>>>(y, gmean, grstd, 512);
    sparsemax_k<<<kB / 8, 256, 0, stream>>>(y, gmean, grstd,
        att_g + (size_t)s * 512, att_b + (size_t)s * 512,
        xn, prior, mxbf, mloss, s == 0 ? 1 : 0);
    run_ft(mxbf, kD, s + 1, out + (size_t)s * kB * 64);
  }
  finalize_k<<<1, 1, 0, stream>>>(mloss, out + (size_t)3 * kB * 64);
}
